// Aligner_50809463111989
// MI455X (gfx1250) — compile-verified, matrix operand order checked by reading, not run on hardware
//
#include <hip/hip_runtime.h>

typedef __attribute__((ext_vector_type(16))) _Float16 v16h;
typedef __attribute__((ext_vector_type(8)))  _Float16 v8h;
typedef __attribute__((ext_vector_type(4)))  _Float16 v4h;
typedef __attribute__((ext_vector_type(8)))  float    v8f;

#define BATCH 16
#define CDIM  512
#define DDIM  512
#define N1DIM 2048
#define N2DIM 2048

#define BM 128
#define BN 128
#define BK 32
#define NTHREADS 256

// ---------------------------------------------------------------------------
// Fragment load from LDS laid out [row][k] (row stride = BK halves = 64B).
// CDNA5 16-bit operand layout: lanes 0-15 hold K=0..7 (VGPR0-3) and K=16..23
// (VGPR4-7); lanes 16-31 hold K=8..15 and K=24..31. Two ds_load_b128 each.
// ---------------------------------------------------------------------------
__device__ __forceinline__ v16h ld_frag_lds(const _Float16* p) {
    v8h lo = *(const v8h*)(p);        // bytes [base, base+16)
    v8h hi = *(const v8h*)(p + 16);   // bytes [base+32, base+48)
    return __builtin_shufflevector(lo, hi, 0,1,2,3,4,5,6,7,8,9,10,11,12,13,14,15);
}

// Stage one K-tile's worth of A/B globals into registers (8x global_load_b128,
// all in flight together — no per-load waits).
template<bool ATRANS>
__device__ __forceinline__ void load_tile_regs(
    const float* __restrict__ A, const float* __restrict__ Bp,
    int tid, int m0, int n0, int lda, int ldb, int k0,
    float4 ra[4], float4 rb[4])
{
    #pragma unroll
    for (int i = 0; i < 4; ++i) {
        int f = tid + i * NTHREADS;              // float4 index, 0..1023
        if (ATRANS) {                            // tile: 32 k-rows x 128 m-cols
            int k = f >> 5, m = (f & 31) << 2;
            ra[i] = *(const float4*)(A + (long)(k0 + k) * lda + m0 + m);
        } else {                                 // tile: 128 m-rows x 32 k-cols
            int m = f >> 3, k = (f & 7) << 2;
            ra[i] = *(const float4*)(A + (long)(m0 + m) * lda + k0 + k);
        }
        int k = f >> 5, n = (f & 31) << 2;       // B tile: 32 k-rows x 128 n-cols
        rb[i] = *(const float4*)(Bp + (long)(k0 + k) * ldb + n0 + n);
    }
}

// Convert f32 -> f16 and scatter into LDS tiles As[m][k], Bs[n][k].
template<bool ATRANS>
__device__ __forceinline__ void store_tile_lds(
    _Float16* As, _Float16* Bs, int tid,
    const float4 ra[4], const float4 rb[4])
{
    #pragma unroll
    for (int i = 0; i < 4; ++i) {
        int f = tid + i * NTHREADS;
        if (ATRANS) {                            // transpose k-major -> [m][k]
            int k = f >> 5, m = (f & 31) << 2;
            As[(m + 0) * BK + k] = (_Float16)ra[i].x;
            As[(m + 1) * BK + k] = (_Float16)ra[i].y;
            As[(m + 2) * BK + k] = (_Float16)ra[i].z;
            As[(m + 3) * BK + k] = (_Float16)ra[i].w;
        } else {                                 // direct [m][k], one ds_store_b64
            int m = f >> 3, k = (f & 7) << 2;
            v4h h = {(_Float16)ra[i].x, (_Float16)ra[i].y,
                     (_Float16)ra[i].z, (_Float16)ra[i].w};
            *(v4h*)(&As[m * BK + k]) = h;
        }
        int k = f >> 5, n = (f & 31) << 2;       // transpose k-major -> [n][k]
        Bs[(n + 0) * BK + k] = (_Float16)rb[i].x;
        Bs[(n + 1) * BK + k] = (_Float16)rb[i].y;
        Bs[(n + 2) * BK + k] = (_Float16)rb[i].z;
        Bs[(n + 3) * BK + k] = (_Float16)rb[i].w;
    }
}

// ---------------------------------------------------------------------------
// C = A * B per batch, f32 in/out, f16 WMMA with f32 accumulation.
// ATRANS=true : A[m][k] = Ag[k*lda + m]   (scores = RI_2^T * RI_1)
// ATRANS=false: A[m][k] = Ag[m*lda + k]   (RE_embed = RE_2 * P)
// B[k][n] = Bg[k*ldb + n] always. Double-buffered LDS pipeline.
// ---------------------------------------------------------------------------
template<bool ATRANS>
__global__ __launch_bounds__(NTHREADS) void gemm_wmma_f16(
    const float* __restrict__ Ag, const float* __restrict__ Bg,
    float* __restrict__ Cg, int K, int lda, int ldb, int ldc,
    long strideA, long strideB, long strideC)
{
    __shared__ _Float16 As[2][BM * BK];   // [m][k]
    __shared__ _Float16 Bs[2][BN * BK];   // [n][k]

    const int tid   = threadIdx.x;
    const int lane  = tid & 31;
    const int wid   = tid >> 5;
    const int waveM = wid >> 2;            // 0..1  (64 rows each)
    const int waveN = wid & 3;             // 0..3  (32 cols each)
    const int lm    = lane & 15;
    const int hiSel = (lane >> 4) & 1;     // 0: lanes 0-15, 1: lanes 16-31

    const int n0 = blockIdx.x * BN;
    const int m0 = blockIdx.y * BM;
    const int b  = blockIdx.z;

    const float* A  = Ag + (long)b * strideA;
    const float* Bp = Bg + (long)b * strideB;
    float*       C  = Cg + (long)b * strideC;

    const v8f vzero = {0.f,0.f,0.f,0.f,0.f,0.f,0.f,0.f};
    v8f acc[4][2];
    #pragma unroll
    for (int t = 0; t < 4; ++t)
        #pragma unroll
        for (int u = 0; u < 2; ++u)
            acc[t][u] = vzero;

    float4 ra[4], rb[4];
    const int NK = K / BK;

    // prologue: tile 0 -> regs -> LDS buf 0
    load_tile_regs<ATRANS>(A, Bp, tid, m0, n0, lda, ldb, 0, ra, rb);
    store_tile_lds<ATRANS>(As[0], Bs[0], tid, ra, rb);
    __syncthreads();

    for (int kt = 0; kt < NK; ++kt) {
        const int  cur     = kt & 1;
        const bool hasNext = (kt + 1) < NK;

        // issue next tile's global loads first; they drain while we do WMMA
        if (hasNext)
            load_tile_regs<ATRANS>(A, Bp, tid, m0, n0, lda, ldb,
                                   (kt + 1) * BK, ra, rb);
        // L2 prefetch two tiles ahead (global_prefetch_b8)
        if (kt + 2 < NK) {
            long k2 = (long)(kt + 2) * BK;
            if (ATRANS)
                __builtin_prefetch(A + (k2 + (tid >> 5)) * lda + m0 + ((tid & 31) << 2), 0, 0);
            else
                __builtin_prefetch(A + (long)(m0 + (tid >> 1)) * lda + k2, 0, 0);
            __builtin_prefetch(Bp + (k2 + (tid >> 5)) * ldb + n0 + ((tid & 31) << 2), 0, 0);
        }

        // fragments + WMMA from buffer `cur`
        v16h afr[4], bfr[2];
        #pragma unroll
        for (int t = 0; t < 4; ++t)
            afr[t] = ld_frag_lds(&As[cur][(waveM * 64 + t * 16 + lm) * BK + hiSel * 8]);
        #pragma unroll
        for (int u = 0; u < 2; ++u)
            bfr[u] = ld_frag_lds(&Bs[cur][(waveN * 32 + u * 16 + lm) * BK + hiSel * 8]);

        #pragma unroll
        for (int t = 0; t < 4; ++t)
            #pragma unroll
            for (int u = 0; u < 2; ++u)
                acc[t][u] = __builtin_amdgcn_wmma_f32_16x16x32_f16(
                    false, afr[t], false, bfr[u], (short)0, acc[t][u], false, false);

        // stash next tile into the other LDS buffer (safe: nobody reads it
        // until after the barrier below)
        if (hasNext)
            store_tile_lds<ATRANS>(As[cur ^ 1], Bs[cur ^ 1], tid, ra, rb);

        __syncthreads();
    }

    // epilogue: CDNA5 C layout — lane 0-15: N=lm, M=r; lane 16-31: N=lm, M=8+r
    #pragma unroll
    for (int t = 0; t < 4; ++t) {
        #pragma unroll
        for (int u = 0; u < 2; ++u) {
            int row0 = m0 + waveM * 64 + t * 16 + hiSel * 8;
            int col  = n0 + waveN * 32 + u * 16 + lm;
            #pragma unroll
            for (int r = 0; r < 8; ++r)
                C[(long)(row0 + r) * ldc + col] = acc[t][u][r];
        }
    }
}

// ---------------------------------------------------------------------------
// Column softmax stats over axis n2: block handles 64 columns, 4 row-groups
// per column, LDS reduction. Writes max and 1/sum(exp) per (b, n1).
// ---------------------------------------------------------------------------
#define SCOLS 64
__global__ __launch_bounds__(256) void colsoftmax_stats(
    const float* __restrict__ S, float* __restrict__ mx, float* __restrict__ rs)
{
    __shared__ float red[4][SCOLS];
    const int cb = blockIdx.x;                    // b*(N1/SCOLS) + group
    const int b  = cb / (N1DIM / SCOLS);
    const int g  = cb % (N1DIM / SCOLS);
    const int c  = threadIdx.x & (SCOLS - 1);     // column within group
    const int rg = threadIdx.x >> 6;              // row-group 0..3
    const int n1 = g * SCOLS + c;
    const float* p = S + (long)b * N2DIM * N1DIM + n1;
    const int rowsPer = N2DIM / 4;                // 512

    float m = -3.402823466e+38f;
    for (int i = rg * rowsPer; i < (rg + 1) * rowsPer; ++i)
        m = fmaxf(m, p[(long)i * N1DIM]);
    red[rg][c] = m;
    __syncthreads();
    const float cm = fmaxf(fmaxf(red[0][c], red[1][c]),
                           fmaxf(red[2][c], red[3][c]));
    __syncthreads();

    float s = 0.f;
    for (int i = rg * rowsPer; i < (rg + 1) * rowsPer; ++i)
        s += __expf(p[(long)i * N1DIM] - cm);
    red[rg][c] = s;
    __syncthreads();
    if (rg == 0) {
        float tot = red[0][c] + red[1][c] + red[2][c] + red[3][c];
        int col = b * N1DIM + n1;
        mx[col] = cm;
        rs[col] = 1.0f / tot;
    }
}

// In-place P = exp(S - max[col]) * rsum[col], float4 vectorized.
__global__ __launch_bounds__(256) void colsoftmax_apply(
    float* __restrict__ S, const float* __restrict__ mx, const float* __restrict__ rs)
{
    long q   = (long)blockIdx.x * blockDim.x + threadIdx.x;  // float4 index
    long idx = q * 4;
    long n1  = idx & (long)(N1DIM - 1);
    long b   = idx / ((long)N1DIM * N2DIM);
    long col = b * N1DIM + n1;
    float4 v  = *(float4*)(S + idx);
    float4 m4 = *(const float4*)(mx + col);
    float4 r4 = *(const float4*)(rs + col);
    v.x = __expf(v.x - m4.x) * r4.x;
    v.y = __expf(v.y - m4.y) * r4.y;
    v.z = __expf(v.z - m4.z) * r4.z;
    v.w = __expf(v.w - m4.w) * r4.w;
    *(float4*)(S + idx) = v;
}

extern "C" void kernel_launch(void* const* d_in, const int* in_sizes, int n_in,
                              void* d_out, int out_size, void* d_ws, size_t ws_size,
                              hipStream_t stream)
{
    const float* RI_1 = (const float*)d_in[0];   // (B, C, N1)
    const float* RI_2 = (const float*)d_in[1];   // (B, C, N2)
    const float* RE_2 = (const float*)d_in[2];   // (B, D, N2)

    float* out      = (float*)d_out;
    float* RE_embed = out;                                   // (B, D, N1)
    float* attn     = out + (long)BATCH * DDIM * N1DIM;      // (B, N2, N1)

    float* mx = (float*)d_ws;                                // B*N1 floats
    float* rs = mx + (long)BATCH * N1DIM;                    // B*N1 floats

    // 1) scores[b,n2,n1] = sum_c RI_2[b,c,n2] * RI_1[b,c,n1]  -> attn region
    dim3 g1(N1DIM / BN, N2DIM / BM, BATCH);
    gemm_wmma_f16<true><<<g1, NTHREADS, 0, stream>>>(
        RI_2, RI_1, attn, CDIM,
        /*lda=*/N2DIM, /*ldb=*/N1DIM, /*ldc=*/N1DIM,
        (long)CDIM * N2DIM, (long)CDIM * N1DIM, (long)N2DIM * N1DIM);

    // 2) per-column (axis n2) softmax statistics
    colsoftmax_stats<<<BATCH * (N1DIM / SCOLS), 256, 0, stream>>>(attn, mx, rs);

    // 3) normalize in place -> attention_map
    long total4 = (long)BATCH * N2DIM * N1DIM / 4;
    colsoftmax_apply<<<(unsigned)(total4 / 256), 256, 0, stream>>>(attn, mx, rs);

    // 4) RE_embed[b,d,n1] = sum_n2 RE_2[b,d,n2] * P[b,n2,n1]
    dim3 g2(N1DIM / BN, DDIM / BM, BATCH);
    gemm_wmma_f16<false><<<g2, NTHREADS, 0, stream>>>(
        RE_2, attn, RE_embed, N2DIM,
        /*lda=*/N2DIM, /*ldb=*/N1DIM, /*ldc=*/N1DIM,
        (long)DDIM * N2DIM, (long)N2DIM * N1DIM, (long)DDIM * N1DIM);
}
